// FeatureGrouper_51616916963756
// MI455X (gfx1250) — matrix operand order repaired and checked by reading, not compiled, not run on hardware
//
#include <hip/hip_runtime.h>

typedef __attribute__((ext_vector_type(16))) __bf16 v16bf;
typedef __attribute__((ext_vector_type(8)))  __bf16 v8bf;
typedef __attribute__((ext_vector_type(8)))  float  v8f;
typedef __attribute__((ext_vector_type(4)))  float  v4f;
typedef __attribute__((ext_vector_type(2)))  float  v2f;

#define NAGG 5
#define EDIM 48
#define QKVD 144
#define ROWF 192
#define OUTF 256
#define NTOK (64 * 4096)

// workspace layout (bf16 element offsets, then a float region)
#define WS_QKVB 0              // 45 tiles * 512 bf16 (fused agg->qkv weights)
#define WS_OUTB 23040          // 6 tiles * 512 bf16 (out_w, K padded 48->64)
#define WS_EXTB 26112          // 1 tile * 512 bf16 (extra_W)
#define WS_BC_BYTES 53248      // 720 floats: fused qkv bias (5 x 144)

// ---------------------------------------------------------------------------
// Prep kernel: fold agg_W into in_proj_w (Wc[a] = in_proj_w @ agg_W[a]),
// convert all weights to bf16 and pre-swizzle into WMMA B-fragment layout:
// tile t, lane l, element j  <->  B[k = (j&7) + 8*(l>>4) + 16*(j>>3)][n = l&15].
// ---------------------------------------------------------------------------
__global__ __launch_bounds__(256) void fg_prep_kernel(
    const float* __restrict__ agg_W,     // (5,48,32)
    const float* __restrict__ agg_b,     // (5,48)
    const float* __restrict__ in_proj_w, // (144,48)
    const float* __restrict__ in_proj_b, // (144,)
    const float* __restrict__ out_w,     // (48,48)
    const float* __restrict__ extra_W,   // (16,32)
    void* __restrict__ ws)
{
    __bf16* wb = (__bf16*)ws;
    float*  bc = (float*)((char*)ws + WS_BC_BYTES);
    const int tid = blockIdx.x * blockDim.x + threadIdx.x;
    const int nth = gridDim.x * blockDim.x;

    // fused qkv weights: Wc[a][jj][k] = sum_e in_proj_w[jj][e] * agg_W[a][e][k]
    for (int u = tid; u < 45 * 512; u += nth) {
        int t = u >> 9, r = u & 511, l = r >> 4, j = r & 15;
        int a = t / 9, c = t % 9;
        int n = l & 15, hb = l >> 4;
        int jj = c * 16 + n;       // output row in [0,144)
        int k  = (j & 7) + 8 * hb + 16 * (j >> 3);      // input col in [0,32)
        float acc = 0.f;
        for (int e = 0; e < 48; ++e)
            acc += in_proj_w[jj * 48 + e] * agg_W[(a * 48 + e) * 32 + k];
        wb[WS_QKVB + u] = (__bf16)acc;
    }
    // out projection weights, B[k][n] = out_w[n][k], K padded 48 -> 64
    for (int u = tid; u < 6 * 512; u += nth) {
        int t = u >> 9, r = u & 511, l = r >> 4, j = r & 15;
        int cn = t >> 1, ck = t & 1;
        int n = l & 15, hb = l >> 4;
        int K = ck * 32 + (j & 7) + 8 * hb + 16 * (j >> 3);
        float v = (K < 48) ? out_w[(cn * 16 + n) * 48 + K] : 0.f;
        wb[WS_OUTB + u] = (__bf16)v;
    }
    // extra weights, B[k][n] = extra_W[n][k] (exactly one 32x16 tile)
    for (int u = tid; u < 512; u += nth) {
        int l = u >> 4, j = u & 15;
        int n = l & 15, hb = l >> 4;
        int k = (j & 7) + 8 * hb + 16 * (j >> 3);
        wb[WS_EXTB + u] = (__bf16)extra_W[n * 32 + k];
    }
    // fused qkv bias: bc[a][jj] = in_proj_w[jj] . agg_b[a] + in_proj_b[jj]
    for (int u = tid; u < 720; u += nth) {
        int a = u / 144, jj = u % 144;
        float acc = in_proj_b[jj];
        for (int e = 0; e < 48; ++e)
            acc += in_proj_w[jj * 48 + e] * agg_b[a * 48 + e];
        bc[u] = acc;
    }
}

// Build a 16-bit A fragment (16x32, K dimension) from 8+8 fp32 values.
// Per the CDNA5 A layout: lanes 0-15 hold K = 0-7 & 16-23, lanes 16-31 hold
// K = 8-15 & 24-31; caller passes lo = row + kbase + 8*half.
// Non-temporal: x is streamed exactly once; keep L2 clean for the weights.
__device__ inline v16bf load_a16_f32_nt(const float* lo)
{
    v4f a0 = __builtin_nontemporal_load((const v4f*)(lo));
    v4f a1 = __builtin_nontemporal_load((const v4f*)(lo + 4));
    v4f b0 = __builtin_nontemporal_load((const v4f*)(lo + 16));
    v4f b1 = __builtin_nontemporal_load((const v4f*)(lo + 20));
    v16bf r;
#pragma unroll
    for (int i = 0; i < 4; ++i) {
        r[i]      = (__bf16)a0[i];
        r[4 + i]  = (__bf16)a1[i];
        r[8 + i]  = (__bf16)b0[i];
        r[12 + i] = (__bf16)b1[i];
    }
    return r;
}

// bf16 pair (packed in one dword) -> float2; bf16->f32 is just "<<16".
__device__ inline v2f bfpair(unsigned int u)
{
    union { unsigned int i; float f; } lo, hi;
    lo.i = u << 16;
    hi.i = u & 0xffff0000u;
    v2f r; r[0] = lo.f; r[1] = hi.f;
    return r;
}

// ---------------------------------------------------------------------------
// Fused main kernel: 1 wave = 16 tokens, 2 waves per block.
// LDS: one bf16 qkv buffer per wave; the attention output o[s] overwrites the
// q[s] slots in-place (q[s] is dead once its scores are computed, identical
// 48-element footprint), so no separate o buffer is needed.
// ---------------------------------------------------------------------------
__global__ __launch_bounds__(64) void fg_fused_kernel(
    const float* __restrict__ x,        // (B,N,192)
    const void*  __restrict__ ws,
    const float* __restrict__ out_b,    // (48,)
    const float* __restrict__ extra_b,  // (16,)
    float* __restrict__ out)            // (B,N,256)
{
    __shared__ __bf16 qkv_s[2][16 * NAGG * QKVD]; // 46080 B total

    const __bf16* wb = (const __bf16*)ws;
    const float*  bc = (const float*)((const char*)ws + WS_BC_BYTES);

    const int lane = threadIdx.x & 31;
    const int w    = threadIdx.x >> 5;
    const int half = lane >> 4;
    const int mn   = lane & 15;           // row (token) for A, col for C/D
    const long tok0 = (long)(blockIdx.x * 2 + w) * 16;
    const float* row = x + (size_t)(tok0 + mn) * ROWF;

    __builtin_prefetch(row, 0, 0);        // global_prefetch_b8

    // ---- Phase 1: fused agg -> QKV projection (45 x v_wmma bf16, K=32) ----
    // unroll kept small so only a couple of B tiles are in flight (VGPR cap)
#pragma unroll 1
    for (int a = 0; a < NAGG; ++a) {
        v16bf A = load_a16_f32_nt(row + a * 32 + 8 * half);
#pragma unroll 3
        for (int c = 0; c < 9; ++c) {
            v16bf Bt = *(const v16bf*)(wb + WS_QKVB + (a * 9 + c) * 512 + lane * 16);
            float bb = bc[a * 144 + c * 16 + mn];
            v8f C;
#pragma unroll
            for (int r = 0; r < 8; ++r) C[r] = bb;
            C = __builtin_amdgcn_wmma_f32_16x16x32_bf16(
                    false, A, false, Bt, (short)0, C, false, false);
#pragma unroll
            for (int r = 0; r < 8; ++r)
                qkv_s[w][((r + 8 * half) * NAGG + a) * QKVD + c * 16 + mn] = (__bf16)C[r];
        }
    }
    __syncthreads();

    // ---- Phase 2: 5x5 attention; 32 lanes = 16 tokens x 2 heads ----
    // bf16-pair dword LDS accesses + float2 math -> v_pk_fma_f32 (packed VALU)
    {
        const int m = mn, h = half;
        const float scale = 0.20412414523193154f; // 1/sqrt(24)
        unsigned int* qw = (unsigned int*)&qkv_s[w][0];
        // dword indices: element offsets are all even (HD=24, 48, 96)
        const int tokbase = m * NAGG * 72;        // token row in dwords (144/2)
        const int hh = h * 12;                    // head offset in dwords
#pragma unroll
        for (int s = 0; s < NAGG; ++s) {
            v2f q2[12];
#pragma unroll
            for (int dd = 0; dd < 12; ++dd)
                q2[dd] = bfpair(qw[tokbase + s * 72 + hh + dd]);
            float sc[NAGG];
#pragma unroll
            for (int t = 0; t < NAGG; ++t) {
                v2f acc = {0.f, 0.f};
#pragma unroll
                for (int dd = 0; dd < 12; ++dd)
                    acc += q2[dd] * bfpair(qw[tokbase + t * 72 + 24 + hh + dd]);
                sc[t] = (acc[0] + acc[1]) * scale;
            }
            float mx = sc[0];
#pragma unroll
            for (int t = 1; t < NAGG; ++t) mx = fmaxf(mx, sc[t]);
            float p[NAGG], sum = 0.f;
#pragma unroll
            for (int t = 0; t < NAGG; ++t) { p[t] = __expf(sc[t] - mx); sum += p[t]; }
            float inv = __builtin_amdgcn_rcpf(sum);
            // write o[s] into the (now dead) q[s] slots, packed bf16 pairs
#pragma unroll
            for (int dd = 0; dd < 12; ++dd) {
                v2f acc = {0.f, 0.f};
#pragma unroll
                for (int t = 0; t < NAGG; ++t)
                    acc += bfpair(qw[tokbase + t * 72 + 48 + hh + dd]) * p[t];
                union { __bf16 b[2]; unsigned int u; } pk;
                pk.b[0] = (__bf16)(acc[0] * inv);
                pk.b[1] = (__bf16)(acc[1] * inv);
                qw[tokbase + s * 72 + hh + dd] = pk.u;
            }
        }
    }
    __syncthreads();

    // ---- Phase 3: output projection (K=48 padded to 64 -> 2 chained WMMAs) ----
    // B tiles are loop-invariant across aggs -> LICM loads them once.
#pragma unroll 1
    for (int a = 0; a < NAGG; ++a) {
        const __bf16* ob = &qkv_s[w][(mn * NAGG + a) * QKVD];   // o aliases q slots
        v16bf A0, A1;
        {
            v8bf lo  = *(const v8bf*)(ob + 8 * half);
            v8bf hi  = *(const v8bf*)(ob + 16 + 8 * half);
            v8bf lo1 = *(const v8bf*)(ob + 32 + 8 * half);
#pragma unroll
            for (int i = 0; i < 8; ++i) {
                A0[i] = lo[i]; A0[8 + i] = hi[i];
                A1[i] = lo1[i]; A1[8 + i] = (__bf16)0.f;  // K >= 48 padding
            }
        }
#pragma unroll
        for (int cn = 0; cn < 3; ++cn) {
            float bb = out_b[cn * 16 + mn];
            v8f C;
#pragma unroll
            for (int r = 0; r < 8; ++r) C[r] = bb;
            v16bf B0 = *(const v16bf*)(wb + WS_OUTB + (cn * 2 + 0) * 512 + lane * 16);
            v16bf B1 = *(const v16bf*)(wb + WS_OUTB + (cn * 2 + 1) * 512 + lane * 16);
            C = __builtin_amdgcn_wmma_f32_16x16x32_bf16(
                    false, A0, false, B0, (short)0, C, false, false);
            C = __builtin_amdgcn_wmma_f32_16x16x32_bf16(
                    false, A1, false, B1, (short)0, C, false, false);
#pragma unroll
            for (int r = 0; r < 8; ++r)
                __builtin_nontemporal_store(
                    C[r], &out[(size_t)(tok0 + r + 8 * half) * OUTF + a * EDIM + cn * 16 + mn]);
        }
    }

    // ---- extra linear: exactly one 16x16x32 WMMA ----
    {
        v16bf Ae = load_a16_f32_nt(row + 160 + 8 * half);
        v16bf Be = *(const v16bf*)(wb + WS_EXTB + lane * 16);
        float bb = extra_b[mn];
        v8f C;
#pragma unroll
        for (int r = 0; r < 8; ++r) C[r] = bb;
        C = __builtin_amdgcn_wmma_f32_16x16x32_bf16(
                false, Ae, false, Be, (short)0, C, false, false);
#pragma unroll
        for (int r = 0; r < 8; ++r)
            __builtin_nontemporal_store(
                C[r], &out[(size_t)(tok0 + r + 8 * half) * OUTF + 240 + mn]);
    }
}

extern "C" void kernel_launch(void* const* d_in, const int* in_sizes, int n_in,
                              void* d_out, int out_size, void* d_ws, size_t ws_size,
                              hipStream_t stream)
{
    (void)in_sizes; (void)n_in; (void)out_size; (void)ws_size;
    const float* x         = (const float*)d_in[0];
    const float* agg_W     = (const float*)d_in[1];
    const float* agg_b     = (const float*)d_in[2];
    const float* in_proj_w = (const float*)d_in[3];
    const float* in_proj_b = (const float*)d_in[4];
    const float* out_w     = (const float*)d_in[5];
    const float* out_b     = (const float*)d_in[6];
    const float* extra_W   = (const float*)d_in[7];
    const float* extra_b   = (const float*)d_in[8];
    float* out = (float*)d_out;

    // 1) build fused / pre-swizzled bf16 weights in workspace (~56 KB)
    fg_prep_kernel<<<32, 256, 0, stream>>>(agg_W, agg_b, in_proj_w, in_proj_b,
                                           out_w, extra_W, d_ws);
    // 2) fused token pipeline: 16 tokens per wave, 2 waves per block
    const int tiles = NTOK / 16;          // 16384
    fg_fused_kernel<<<tiles / 2, 64, 0, stream>>>(x, d_ws, out_b, extra_b, out);
}
